// LSTM_15556371546824
// MI455X (gfx1250) — matrix operand order repaired and checked by reading, not compiled, not run on hardware
//
#include <hip/hip_runtime.h>
#include <hip/hip_bf16.h>

// ---------------- problem constants ----------------
#define BSZ   32
#define TLEN  2048
#define DD    512
#define HH    512
#define GG    2048              // 4*H gates
#define KTOT  1024              // D + H (fused A = [x_t ; h_{t-1}])
#define NBLK  8                 // persistent workgroups (each owns 64 hidden cols x 4 gates)

typedef _Float16 h8  __attribute__((ext_vector_type(8)));
typedef _Float16 h16 __attribute__((ext_vector_type(16)));
typedef float    f8  __attribute__((ext_vector_type(8)));

__device__ __forceinline__ float sigf(float x) {
    return 1.0f / (1.0f + __expf(-x));
}
__device__ __forceinline__ float tanh_fast(float x) {
    float e = __expf(2.0f * x);
    return (e - 1.0f) / (e + 1.0f);
}

// ---------------- prep: pack [w_ih | w_hh] -> f16, bias sum, h0 copy ----------------
__global__ void lstm_prep_kernel(const float* __restrict__ w_ih,
                                 const float* __restrict__ w_hh,
                                 const float* __restrict__ b_ih,
                                 const float* __restrict__ b_hh,
                                 const float* __restrict__ hx,
                                 _Float16* __restrict__ wc,   // [G][KTOT] f16
                                 float*    __restrict__ bias, // [G]
                                 float*    __restrict__ h0)   // [B][H]
{
    const size_t WCN  = (size_t)GG * KTOT;
    const size_t tot  = WCN + GG + (size_t)BSZ * HH;
    for (size_t idx = (size_t)blockIdx.x * blockDim.x + threadIdx.x;
         idx < tot; idx += (size_t)gridDim.x * blockDim.x) {
        if (idx < WCN) {
            int n = (int)(idx >> 10);
            int k = (int)(idx & 1023);
            float v = (k < DD) ? w_ih[n * DD + k] : w_hh[n * HH + (k - DD)];
            wc[idx] = (_Float16)v;
        } else if (idx < WCN + GG) {
            int n = (int)(idx - WCN);
            bias[n] = b_ih[n] + b_hh[n];
        } else {
            int i = (int)(idx - WCN - GG);
            h0[i] = hx[i];
        }
    }
}

// ---------------- persistent recurrent kernel ----------------
__global__ __launch_bounds__(256, 1)
void lstm_rec_kernel(const float* __restrict__ x,     // [B][T][D]
                     const float* __restrict__ cx,    // [B][H]
                     const _Float16* __restrict__ wc, // [G][KTOT]
                     const float* __restrict__ bias,  // [G]
                     float* __restrict__ hbuf,        // 2 x [B][H] double buffer
                     unsigned* __restrict__ cnt,      // grid barrier counter
                     float* __restrict__ fg_out,      // [B][T][H]
                     float* __restrict__ h_out,       // [B][T][H]
                     float* __restrict__ g_out,       // [B][T][H]
                     float* __restrict__ o_out)       // [B][H]
{
    // LDS A-stage: 32 rows x 512 K-cols (one K-half per phase), padded row
    __shared__ _Float16 Ash[32][520];   // 33,280 B

    const int tid   = threadIdx.x;
    const int blk   = blockIdx.x;       // 0..7 -> hidden cols [blk*64, blk*64+64)
    const int wave  = tid >> 5;         // 0..7
    const int lane  = tid & 31;
    const int half  = lane >> 4;
    const int l15   = lane & 15;
    const int mt    = wave >> 2;        // 0..1 (M tile of 16)
    const int chunk = wave & 3;         // 0..3 (16-col hidden chunk)
    const int m_base = mt * 16;
    const int hcol   = blk * 64 + chunk * 16 + l15;

    // cell state lives in registers for the whole sequence
    float creg[8];
#pragma unroll
    for (int e = 0; e < 8; ++e) {
        int m = m_base + e + 8 * half;
        creg[e] = cx[m * HH + hcol];
    }
    float bgate[4];
#pragma unroll
    for (int g = 0; g < 4; ++g) bgate[g] = bias[g * HH + hcol];

    const size_t BTHs = (size_t)TLEN * HH;

    for (int t = 0; t < TLEN; ++t) {
        const float* hcur  = hbuf + (size_t)(t & 1) * (BSZ * HH);
        float*       hnext = hbuf + (size_t)((t + 1) & 1) * (BSZ * HH);

        f8 acc[4];
#pragma unroll
        for (int g = 0; g < 4; ++g) {
            f8 z = {0.f, 0.f, 0.f, 0.f, 0.f, 0.f, 0.f, 0.f};
            acc[g] = z;
        }

#pragma unroll 1
        for (int p = 0; p < 2; ++p) {
            // ---- stage A half-K: p==0 -> x_t (fp32->f16), p==1 -> h_{t-1} ----
            for (int i = tid; i < 32 * 128; i += 256) {
                int r  = i >> 7;
                int c4 = (i & 127) << 2;
                const float* src = (p == 0)
                    ? (x + ((size_t)r * TLEN + (size_t)t) * DD + c4)
                    : (hcur + r * HH + c4);
                float4 v = *(const float4*)src;
                _Float16* dst = &Ash[r][c4];
                dst[0] = (_Float16)v.x; dst[1] = (_Float16)v.y;
                dst[2] = (_Float16)v.z; dst[3] = (_Float16)v.w;
            }
            __syncthreads();

            // ---- K loop over this 512-wide half (16 x k=32) ----
#pragma unroll 2
            for (int kk = 0; kk < 16; ++kk) {
                const int k0 = kk * 32;
                const int m  = m_base + l15;
                // A fragment: contiguous k = 16*half + s, two 16B chunks per lane
                const h8 a0 = *(const h8*)&Ash[m][k0 + half * 16];
                const h8 a1 = *(const h8*)&Ash[m][k0 + half * 16 + 8];
                const h16 afrag = __builtin_shufflevector(
                    a0, a1, 0, 1, 2, 3, 4, 5, 6, 7, 8, 9, 10, 11, 12, 13, 14, 15);

                const int kg = p * 512 + k0 + half * 16;  // global K index
#pragma unroll
                for (int g = 0; g < 4; ++g) {
                    const int n = g * HH + blk * 64 + chunk * 16 + l15;
                    const _Float16* wp = wc + (size_t)n * KTOT + kg;
                    // prefetch next k-iteration's weights into L0/L2
                    __builtin_prefetch(wp + 32, 0, 1);
                    const h16 bfrag = *(const h16*)wp;
                    acc[g] = __builtin_amdgcn_wmma_f32_16x16x32_f16(
                        false, afrag, false, bfrag, (short)0, acc[g], false, false);
                }
            }
            __syncthreads();  // protect Ash before restage
        }

        // ---- fused LSTM cell elementwise (all 4 gates in-register) ----
#pragma unroll
        for (int e = 0; e < 8; ++e) {
            const float iv = sigf(acc[0][e] + bgate[0]);
            const float fv = sigf(acc[1][e] + bgate[1]);
            const float gv = tanh_fast(acc[2][e] + bgate[2]);
            const float ov = sigf(acc[3][e] + bgate[3]);
            const float cn = fv * creg[e] + iv * gv;
            creg[e] = cn;
            const float hn = ov * tanh_fast(cn);

            const int m = m_base + e + 8 * half;
            const size_t oidx = (size_t)m * BTHs + (size_t)t * HH + hcol;
            fg_out[oidx] = fv;
            h_out[oidx]  = hn;
            g_out[oidx]  = gv;
            hnext[m * HH + hcol] = hn;
            if (t == TLEN - 1) o_out[m * HH + hcol] = ov;
        }

        // ---- grid barrier (release h writes, acquire before next read) ----
        __syncthreads();
        if (tid == 0) {
            __hip_atomic_fetch_add(cnt, 1u, __ATOMIC_RELEASE, __HIP_MEMORY_SCOPE_AGENT);
            const unsigned target = (unsigned)(t + 1) * NBLK;
            while (__hip_atomic_load(cnt, __ATOMIC_ACQUIRE, __HIP_MEMORY_SCOPE_AGENT) < target) {
                __builtin_amdgcn_s_sleep(1);
            }
        }
        __syncthreads();
    }
}

// ---------------- host launcher ----------------
extern "C" void kernel_launch(void* const* d_in, const int* in_sizes, int n_in,
                              void* d_out, int out_size, void* d_ws, size_t ws_size,
                              hipStream_t stream) {
    const float* x    = (const float*)d_in[0];
    const float* hx   = (const float*)d_in[1];
    const float* cx   = (const float*)d_in[2];
    const float* w_ih = (const float*)d_in[3];
    const float* w_hh = (const float*)d_in[4];
    const float* b_ih = (const float*)d_in[5];
    const float* b_hh = (const float*)d_in[6];

    // workspace layout (~4.4 MB total)
    char* ws = (char*)d_ws;
    const size_t WC_BYTES   = (size_t)GG * KTOT * sizeof(_Float16); // 4 MB
    const size_t BIAS_BYTES = (size_t)GG * sizeof(float);           // 8 KB
    const size_t HBUF_BYTES = (size_t)2 * BSZ * HH * sizeof(float); // 128 KB
    _Float16* wc   = (_Float16*)ws;
    float*    bias = (float*)(ws + WC_BYTES);
    float*    hbuf = (float*)(ws + WC_BYTES + BIAS_BYTES);
    unsigned* cnt  = (unsigned*)(ws + WC_BYTES + BIAS_BYTES + HBUF_BYTES);

    float* out = (float*)d_out;
    const size_t BTH = (size_t)BSZ * TLEN * HH;
    float* fg = out;
    float* hs = out + BTH;
    float* gs = out + 2 * BTH;
    float* ol = out + 3 * BTH;

    // reset the grid-barrier counter every launch (graph-capture safe)
    hipMemsetAsync(cnt, 0, 256, stream);

    lstm_prep_kernel<<<2048, 256, 0, stream>>>(w_ih, w_hh, b_ih, b_hh, hx,
                                               wc, bias, hbuf);
    lstm_rec_kernel<<<NBLK, 256, 0, stream>>>(x, cx, wc, bias, hbuf, cnt,
                                              fg, hs, gs, ol);
}
